// Is_Stable_GNN_54202487275958
// MI455X (gfx1250) — matrix operand order repaired and checked by reading, not compiled, not run on hardware
//
#include <hip/hip_runtime.h>
#include <hip/hip_fp16.h>

#define NNODES 50000
#define HEADS 16
#define CH 8
#define HCH 128          // HEADS*CH
#define NEDGES 800000
#define ETOT (NEDGES + NNODES)
#define NGRAPH 128
#define NEG_SLOPE 0.2f
#define BN_EPS 1e-5f

typedef __attribute__((ext_vector_type(16))) _Float16 v16h;
typedef __attribute__((ext_vector_type(8)))  _Float16 v8h;
typedef __attribute__((ext_vector_type(8)))  float    v8f;

// ---------- helpers ----------
__device__ __forceinline__ unsigned f2ord(float f) {
    unsigned u = __float_as_uint(f);
    return (u & 0x80000000u) ? ~u : (u | 0x80000000u);
}
__device__ __forceinline__ float ord2f(unsigned u) {
    u = (u & 0x80000000u) ? (u & 0x7FFFFFFFu) : ~u;
    return __uint_as_float(u);
}
__device__ __forceinline__ int edge_src(const int* ei, int e) {
    return (e < NEDGES) ? ei[e] : (e - NEDGES);
}
__device__ __forceinline__ int edge_dst(const int* ei, int e) {
    return (e < NEDGES) ? ei[NEDGES + e] : (e - NEDGES);
}
__device__ __forceinline__ float lrelu(float v) {
    return v > 0.0f ? v : NEG_SLOPE * v;
}

// ---------- utility kernels ----------
__global__ void fill_f32_k(float* p, float v, int n) {
    int i = blockIdx.x * blockDim.x + threadIdx.x;
    if (i < n) p[i] = v;
}
__global__ void fill_u32_k(unsigned* p, unsigned v, int n) {
    int i = blockIdx.x * blockDim.x + threadIdx.x;
    if (i < n) p[i] = v;
}
__global__ void cvt_f16_k(const float* __restrict__ in, _Float16* __restrict__ out, int n) {
    int i = blockIdx.x * blockDim.x + threadIdx.x;
    if (i < n) out[i] = (_Float16)in[i];
}
// W[k, n] row-major (k = input feature, n = output channel) -> Wt[n, k] f16
__global__ void wtrans_k(const float* __restrict__ W, _Float16* __restrict__ Wt) {
    int i = blockIdx.x * blockDim.x + threadIdx.x;
    if (i < HCH * HCH) {
        int k = i >> 7, n = i & 127;
        Wt[n * HCH + k] = (_Float16)W[i];
    }
}

// ---------- WMMA GEMM: D[N,128] = A[N,128](f16) @ Wt^T ; Wt is [n,k] ----------
__global__ __launch_bounds__(256)
void wmma_gemm_128(const _Float16* __restrict__ A, const _Float16* __restrict__ Bt,
                   float* __restrict__ D) {
    const int rowBase = blockIdx.x * 16;
    const int wave = threadIdx.x >> 5;      // 8 waves -> 8 column tiles
    const int lane = threadIdx.x & 31;
    const int l16  = lane & 15;
    const int hh   = lane >> 4;             // half-wave selector
    const int n    = wave * 16 + l16;       // output column

    v8f acc = {};
    const _Float16* arow = A  + (size_t)(rowBase + l16) * HCH;
    const _Float16* bcol = Bt + (size_t)n * HCH;

    #pragma unroll
    for (int kb = 0; kb < 4; ++kb) {        // K = 128 = 4 x 32
        // A 16x32 f16 layout: lane holds row M=l16; halves hh*8..+7 and 16+hh*8..+7
        const _Float16* ap = arow + kb * 32 + hh * 8;
        v8h a0 = *(const v8h*)(ap);
        v8h a1 = *(const v8h*)(ap + 16);
        v16h a, b;
        #pragma unroll
        for (int i = 0; i < 8; ++i) { a[i] = a0[i]; a[8 + i] = a1[i]; }
        // B 32x16 f16: lane holds column n; halves hh*8..+7 and 16+hh*8..+7 (same K order as A)
        const _Float16* bp = bcol + kb * 32 + hh * 8;
        v8h b0 = *(const v8h*)(bp);
        v8h b1 = *(const v8h*)(bp + 16);
        #pragma unroll
        for (int i = 0; i < 8; ++i) { b[i] = b0[i]; b[8 + i] = b1[i]; }
        acc = __builtin_amdgcn_wmma_f32_16x16x32_f16(
            /*neg_a=*/false, a, /*neg_b=*/false, b,
            /*c_mod=*/(short)0, acc, /*reuse_a=*/false, /*reuse_b=*/false);
    }
    // D 16x16 f32 layout: VGPR r -> M = hh*8 + r, column = n
    #pragma unroll
    for (int r = 0; r < 8; ++r)
        D[(size_t)(rowBase + hh * 8 + r) * HCH + n] = acc[r];
}

// ---------- attention projections: a_s[n,h] = <h[n,h,:], asrc[h,:]> ----------
__global__ void attn_k(const float* __restrict__ H, const float* __restrict__ asrc,
                       const float* __restrict__ adst,
                       float* __restrict__ As, float* __restrict__ Ad) {
    int i = blockIdx.x * blockDim.x + threadIdx.x;   // n*HEADS + h
    if (i >= NNODES * HEADS) return;
    int h = i & 15;
    const float* hp = H + (size_t)(i >> 4) * HCH + h * CH;
    const float* sp = asrc + h * CH;
    const float* dp = adst + h * CH;
    float s = 0.f, d = 0.f;
    #pragma unroll
    for (int c = 0; c < CH; ++c) { s += hp[c] * sp[c]; d += hp[c] * dp[c]; }
    As[i] = s; Ad[i] = d;
}

// ---------- edge pass 1: per-destination segment max ----------
__global__ void edge_max_k(const int* __restrict__ ei, const float* __restrict__ As,
                           const float* __restrict__ Ad, unsigned* __restrict__ Emax) {
    int i = blockIdx.x * blockDim.x + threadIdx.x;
    if (i >= ETOT * HEADS) return;
    int h = i & 15, e = i >> 4;
    int s = edge_src(ei, e), d = edge_dst(ei, e);
    float v = lrelu(As[s * HEADS + h] + Ad[d * HEADS + h]);
    atomicMax(&Emax[d * HEADS + h], f2ord(v));
}

// ---------- edge pass 2: softmax denominator ----------
__global__ void edge_den_k(const int* __restrict__ ei, const float* __restrict__ As,
                           const float* __restrict__ Ad, const unsigned* __restrict__ Emax,
                           float* __restrict__ Den) {
    int i = blockIdx.x * blockDim.x + threadIdx.x;
    if (i >= ETOT * HEADS) return;
    int h = i & 15, e = i >> 4;
    int s = edge_src(ei, e), d = edge_dst(ei, e);
    float v = lrelu(As[s * HEADS + h] + Ad[d * HEADS + h]);
    float ex = __expf(v - ord2f(Emax[d * HEADS + h]));
    atomicAdd(&Den[d * HEADS + h], ex);
}

// ---------- edge pass 3: alpha-weighted scatter-add of h[src] ----------
__global__ void edge_agg_k(const int* __restrict__ ei, const float* __restrict__ As,
                           const float* __restrict__ Ad, const unsigned* __restrict__ Emax,
                           const float* __restrict__ Den, const float* __restrict__ H,
                           float* __restrict__ Out) {
    int i = blockIdx.x * blockDim.x + threadIdx.x;
    if (i >= ETOT * HEADS) return;
    int h = i & 15, e = i >> 4;
    int s = edge_src(ei, e), d = edge_dst(ei, e);
    float v = lrelu(As[s * HEADS + h] + Ad[d * HEADS + h]);
    float alpha = __expf(v - ord2f(Emax[d * HEADS + h])) / (Den[d * HEADS + h] + 1e-16f);
    const float* hs = H + (size_t)s * HCH + h * CH;
    float* od = Out + (size_t)d * HCH + h * CH;
    #pragma unroll
    for (int c = 0; c < CH; ++c) atomicAdd(&od[c], hs[c] * alpha);
}

// ---------- BN statistics: per-channel sum / sumsq over all nodes ----------
__global__ void bn_stats_k(const float* __restrict__ X, float* __restrict__ Sum,
                           float* __restrict__ Sq) {
    __shared__ float ss[256], sq[256];
    int c = threadIdx.x & 127;
    float s = 0.f, q = 0.f;
    size_t stride = (size_t)gridDim.x * blockDim.x;  // multiple of 128
    for (size_t i = (size_t)blockIdx.x * blockDim.x + threadIdx.x;
         i < (size_t)NNODES * HCH; i += stride) {
        float x = X[i]; s += x; q += x * x;
    }
    ss[threadIdx.x] = s; sq[threadIdx.x] = q;
    __syncthreads();
    if (threadIdx.x < 128) {
        atomicAdd(&Sum[c], ss[threadIdx.x] + ss[threadIdx.x + 128]);
        atomicAdd(&Sq[c],  sq[threadIdx.x] + sq[threadIdx.x + 128]);
    }
}

// ---------- BN + ReLU; write f32 activation and f16 copy (next GEMM A) ----------
__global__ void bn_apply_k(const float* __restrict__ X, const float* __restrict__ Sum,
                           const float* __restrict__ Sq, const float* __restrict__ G,
                           const float* __restrict__ Beta,
                           float* __restrict__ Yf, _Float16* __restrict__ Yh) {
    int i = blockIdx.x * blockDim.x + threadIdx.x;
    if (i >= NNODES * HCH) return;
    int c = i & 127;
    const float inv = 1.0f / (float)NNODES;
    float mu = Sum[c] * inv;
    float var = Sq[c] * inv - mu * mu;
    float y = (X[i] - mu) * rsqrtf(var + BN_EPS) * G[c] + Beta[c];
    y = fmaxf(y, 0.f);
    Yf[i] = y;
    Yh[i] = (_Float16)y;
}

// ---------- graph pooling ----------
__global__ void pool_k(const float* __restrict__ H, const int* __restrict__ batch,
                       float* __restrict__ Pooled) {
    int i = blockIdx.x * blockDim.x + threadIdx.x;
    if (i >= NNODES * HCH) return;
    int n = i >> 7, c = i & 127;
    atomicAdd(&Pooled[batch[n] * HCH + c], H[i]);
}
__global__ void cnt_k(const int* __restrict__ batch, float* __restrict__ Cnt) {
    int n = blockIdx.x * blockDim.x + threadIdx.x;
    if (n < NNODES) atomicAdd(&Cnt[batch[n]], 1.0f);
}

// ---------- final linear + BN over graphs (single block) ----------
__global__ void head_k(const float* __restrict__ Pooled, const float* __restrict__ Cnt,
                       const float* __restrict__ Wl, const float* __restrict__ G4,
                       const float* __restrict__ Beta4, float* __restrict__ Out) {
    __shared__ float zs[NGRAPH * 2];
    __shared__ float stat[4];                 // mu0, mu1, var0, var1
    int t = threadIdx.x;                      // 256 threads
    int g = t >> 1, cls = t & 1;
    float cn = fmaxf(Cnt[g], 1.0f);
    const float* ps = Pooled + g * HCH;
    float z = 0.f;
    for (int j = 0; j < HCH; ++j) {
        z += (ps[j] / cn) * Wl[j * 2 + cls];          // pooled mean part
        z += ps[j] * Wl[(HCH + j) * 2 + cls];         // pooled sum part
    }
    zs[t] = z;
    __syncthreads();
    if (t < 2) {
        float s = 0.f, q = 0.f;
        for (int gg = 0; gg < NGRAPH; ++gg) { float v = zs[gg * 2 + t]; s += v; q += v * v; }
        float mu = s / (float)NGRAPH;
        stat[t] = mu; stat[2 + t] = q / (float)NGRAPH - mu * mu;
    }
    __syncthreads();
    Out[t] = (z - stat[cls]) * rsqrtf(stat[2 + cls] + BN_EPS) * G4[cls] + Beta4[cls];
}

// ---------- launcher ----------
extern "C" void kernel_launch(void* const* d_in, const int* in_sizes, int n_in,
                              void* d_out, int out_size, void* d_ws, size_t ws_size,
                              hipStream_t stream) {
    (void)in_sizes; (void)n_in; (void)out_size; (void)ws_size;
    const float* x     = (const float*)d_in[0];
    const int*   ei    = (const int*)d_in[1];
    const int*   batch = (const int*)d_in[2];

    char* w = (char*)d_ws;
    auto alloc = [&](size_t b) { char* p = w; w += (b + 255) & ~(size_t)255; return p; };
    _Float16* Af16  = (_Float16*)alloc((size_t)NNODES * HCH * 2);   // GEMM A (f16)
    _Float16* Wt    = (_Float16*)alloc((size_t)HCH * HCH * 2);      // W^T (f16)
    float*    Hg    = (float*)alloc((size_t)NNODES * HCH * 4);      // GEMM out / activ
    float*    Hagg  = (float*)alloc((size_t)NNODES * HCH * 4);      // aggregated
    float*    As    = (float*)alloc((size_t)NNODES * HEADS * 4);
    float*    Ad    = (float*)alloc((size_t)NNODES * HEADS * 4);
    unsigned* Emax  = (unsigned*)alloc((size_t)NNODES * HEADS * 4);
    float*    Den   = (float*)alloc((size_t)NNODES * HEADS * 4);
    float*    Bsum  = (float*)alloc(HCH * 4);
    float*    Bsq   = (float*)alloc(HCH * 4);
    float*    Pool  = (float*)alloc((size_t)NGRAPH * HCH * 4);
    float*    Cnt   = (float*)alloc(NGRAPH * 4);

    const int NC = NNODES * HCH;                 // 6.4M
    const int NH = NNODES * HEADS;               // 800K
    const int EH = ETOT * HEADS;                 // 13.6M
    const int B = 256;

    cvt_f16_k<<<(NC + B - 1) / B, B, 0, stream>>>(x, Af16, NC);

    for (int L = 0; L < 3; ++L) {
        const float* W    = (const float*)d_in[3 + L * 6];
        const float* asrc = (const float*)d_in[5 + L * 6];
        const float* adst = (const float*)d_in[6 + L * 6];
        const float* gg   = (const float*)d_in[7 + L * 6];
        const float* bb   = (const float*)d_in[8 + L * 6];
        // (GAT bias b and final bl cancel inside training-mode BN — skipped.)

        wtrans_k<<<(HCH * HCH + B - 1) / B, B, 0, stream>>>(W, Wt);
        wmma_gemm_128<<<NNODES / 16, B, 0, stream>>>(Af16, Wt, Hg);
        attn_k<<<(NH + B - 1) / B, B, 0, stream>>>(Hg, asrc, adst, As, Ad);

        fill_f32_k<<<(NC + B - 1) / B, B, 0, stream>>>(Hagg, 0.f, NC);
        fill_u32_k<<<(NH + B - 1) / B, B, 0, stream>>>(Emax, 0u, NH);
        fill_f32_k<<<(NH + B - 1) / B, B, 0, stream>>>(Den, 0.f, NH);

        edge_max_k<<<(EH + B - 1) / B, B, 0, stream>>>(ei, As, Ad, Emax);
        edge_den_k<<<(EH + B - 1) / B, B, 0, stream>>>(ei, As, Ad, Emax, Den);
        edge_agg_k<<<(EH + B - 1) / B, B, 0, stream>>>(ei, As, Ad, Emax, Den, Hg, Hagg);

        fill_f32_k<<<1, B, 0, stream>>>(Bsum, 0.f, HCH);
        fill_f32_k<<<1, B, 0, stream>>>(Bsq, 0.f, HCH);
        bn_stats_k<<<1024, B, 0, stream>>>(Hagg, Bsum, Bsq);
        bn_apply_k<<<(NC + B - 1) / B, B, 0, stream>>>(Hagg, Bsum, Bsq, gg, bb, Hg, Af16);
        (void)bb;
    }

    fill_f32_k<<<(NGRAPH * HCH + B - 1) / B, B, 0, stream>>>(Pool, 0.f, NGRAPH * HCH);
    fill_f32_k<<<1, NGRAPH, 0, stream>>>(Cnt, 0.f, NGRAPH);
    pool_k<<<(NC + B - 1) / B, B, 0, stream>>>(Hg, batch, Pool);
    cnt_k<<<(NNODES + B - 1) / B, B, 0, stream>>>(batch, Cnt);
    head_k<<<1, 256, 0, stream>>>(Pool, Cnt, (const float*)d_in[21],
                                  (const float*)d_in[23], (const float*)d_in[24],
                                  (float*)d_out);
}
